// EdgeDecoder_55190329753794
// MI455X (gfx1250) — matrix operand order repaired and checked by reading, not compiled, not run on hardware
//
#include <hip/hip_runtime.h>

typedef __attribute__((ext_vector_type(16))) _Float16 v16h;
typedef __attribute__((ext_vector_type(8)))  float    v8f;
typedef __attribute__((ext_vector_type(4)))  unsigned u32x4;

// ---------------- geometry ----------------
#define TILE_E     64          // edges per block
#define NODE_DIM   128
#define HIDDEN     128
#define IN_DIM     545         // 4*128 + 1 + 32
#define K1_PAD     576         // 18 k-tiles of 32
#define KP1        288         // K1_PAD/2 packed f16 pairs
#define KP2        64          // HIDDEN/2
#define XS_STRIDE  584         // halves per X row (16B aligned, 16-row bank-conflict free)
#define XH_STRIDE  136         // halves per h1 row
#define LDS_XS_B   (TILE_E * XS_STRIDE * 2)                 // 74752
#define LDS_XH_B   (TILE_E * XH_STRIDE * 2)                 // 17408
#define LDS_BIAS_B (3 * 128 * 4)                            // b1, b2, W3
#define LDS_TOTAL  (LDS_XS_B + LDS_XH_B + LDS_BIAS_B)       // 93696

union Frag { u32x4 q[2]; v16h h; };

__device__ __forceinline__ unsigned pk2h(float a, float b) {
    union { _Float16 h[2]; unsigned u; } t;
    t.h[0] = (_Float16)a; t.h[1] = (_Float16)b;
    return t.u;
}

// -------- prep: W1[545x128],W2[128x128] f32 -> packed f16 B-fragment layout --------
// w1pt[n][kp] = pack(W1[2kp][n], W1[2kp+1][n]), kp in [0,288), zero-padded past 545
// w2pt[n][kp] = pack(W2[2kp][n], W2[2kp+1][n]), kp in [0,64)
__global__ __launch_bounds__(128) void pack_weights(const float* __restrict__ W1,
                                                    const float* __restrict__ W2,
                                                    unsigned* __restrict__ w1pt,
                                                    unsigned* __restrict__ w2pt) {
    const int n  = threadIdx.x;          // 0..127
    const int kb = blockIdx.x;           // 0..351
    if (kb < KP1) {
        const int k0 = 2 * kb, k1 = 2 * kb + 1;
        const float a = (k0 < IN_DIM) ? W1[(size_t)k0 * HIDDEN + n] : 0.0f;
        const float b = (k1 < IN_DIM) ? W1[(size_t)k1 * HIDDEN + n] : 0.0f;
        w1pt[(size_t)n * KP1 + kb] = pk2h(a, b);
    } else {
        const int kp = kb - KP1;         // 0..63
        const float a = W2[(size_t)(2 * kp)     * HIDDEN + n];
        const float b = W2[(size_t)(2 * kp + 1) * HIDDEN + n];
        w2pt[(size_t)n * KP2 + kp] = pk2h(a, b);
    }
}

// ---------------- main fused edge-MLP kernel ----------------
__global__ __launch_bounds__(128) void edge_mlp(
    const float* __restrict__ z,          // [N_NODES][128]
    const int*   __restrict__ srcIdx,     // [nE]
    const int*   __restrict__ dstIdx,     // [nE]
    const float* __restrict__ attr,       // [nE][7]
    const float* __restrict__ lre,        // [2000][32]
    const float* __restrict__ b1,
    const float* __restrict__ b2,
    const float* __restrict__ W3,         // [128]
    const float* __restrict__ b3,
    const unsigned* __restrict__ w1pt,    // [128][288]
    const unsigned* __restrict__ w2pt,    // [128][64]
    float* __restrict__ out, int nE)
{
    extern __shared__ char smem[];
    _Float16* Xs  = (_Float16*)smem;                    // [64][584] f16 feature tile
    _Float16* Xh1 = (_Float16*)(smem + LDS_XS_B);       // [64][136] f16 h1 tile
    float*    b1s = (float*)(smem + LDS_XS_B + LDS_XH_B);
    float*    b2s = b1s + 128;
    float*    W3s = b2s + 128;
    unsigned* XsU = (unsigned*)Xs;
    unsigned* XhU = (unsigned*)Xh1;

    const int tid  = threadIdx.x;
    const int lane = tid & 31;
    const int w    = tid >> 5;            // wave 0..3, owns edge rows [16w,16w+16)
    const int e0   = (int)blockIdx.x * TILE_E;

    // ---- Phase A: build X tile [64][576] in LDS (f16) ----
    {
        const int cp = tid & 63;          // column pair 0..63 (covers 128 dims)
        const int c  = cp << 1;
        for (int r = tid >> 6; r < TILE_E; r += 2) {
            int e = e0 + r; if (e >= nE) e = nE - 1;
            const int s = srcIdx[e];
            const int d = dstIdx[e];
            const float2 zs = *(const float2*)(z + (size_t)s * NODE_DIM + c);
            const float2 zd = *(const float2*)(z + (size_t)d * NODE_DIM + c);
            unsigned* row = XsU + r * (XS_STRIDE / 2);
            row[cp]        = pk2h(zs.x, zs.y);                              // zs
            row[64  + cp]  = pk2h(zd.x, zd.y);                              // zd
            row[128 + cp]  = pk2h(zs.x * zd.x, zs.y * zd.y);                // zs*zd
            row[192 + cp]  = pk2h(fabsf(zs.x - zd.x), fabsf(zs.y - zd.y)); // |zs-zd|
        }
    }
    if (tid < TILE_E) {
        const int r = tid;
        int e = e0 + r; if (e >= nE) e = nE - 1;
        _Float16* row = Xs + r * XS_STRIDE;
        row[512] = (_Float16)attr[(size_t)e * 7 + 0];            // dist
        const int lrid = (int)attr[(size_t)e * 7 + 2];
        const float* lv = lre + (size_t)lrid * 32;
        #pragma unroll
        for (int j = 0; j < 32; ++j) row[513 + j] = (_Float16)lv[j];
        #pragma unroll
        for (int j = IN_DIM; j < K1_PAD; ++j) row[j] = (_Float16)0.0f;
    }
    if (tid < 128) { b1s[tid] = b1[tid]; b2s[tid] = b2[tid]; W3s[tid] = W3[tid]; }
    __syncthreads();

    const int m    = lane & 15;           // M row within wave tile / N col within n-tile
    const int hi   = lane >> 4;           // lane half selects K sub-range
    const int rowA = 16 * w + m;

    // ---- GEMM1: h1[64x128] = relu(X @ W1 + b1), K = 576 = 18 tiles ----
    const v8f vzero = {0.f, 0.f, 0.f, 0.f, 0.f, 0.f, 0.f, 0.f};
    v8f acc[8];
    #pragma unroll
    for (int nt = 0; nt < 8; ++nt) acc[nt] = vzero;

    for (int kt = 0; kt < 18; ++kt) {
        // A fragment: 2x ds_load_b128
        Frag a;
        const unsigned* ap = XsU + rowA * (XS_STRIDE / 2) + kt * 16 + hi * 4;
        a.q[0] = *(const u32x4*)(ap);       // K pairs v=0..3
        a.q[1] = *(const u32x4*)(ap + 8);   // K pairs v=4..7 (halves +16)
        if (kt + 1 < 18)                    // gfx1250 global_prefetch of next W1 k-tile
            __builtin_prefetch(w1pt + (size_t)m * KP1 + (kt + 1) * 16, 0, 3);
        // Preload ALL 8 B fragments as one 16-load clause.
        Frag bf[8];
        #pragma unroll
        for (int nt = 0; nt < 8; ++nt) {
            const unsigned* bp = w1pt + (size_t)(nt * 16 + m) * KP1 + kt * 16 + hi * 4;
            bf[nt].q[0] = *(const u32x4*)(bp);
            bf[nt].q[1] = *(const u32x4*)(bp + 8);
        }
        // Hard scheduling fence: keep all 16 loads ahead of the WMMA burst so
        // the waits come out staggered (loadcnt 14,12,...) instead of a full
        // s_wait_loadcnt 0 before every WMMA.
        __builtin_amdgcn_sched_barrier(0);
        #pragma unroll
        for (int nt = 0; nt < 8; ++nt) {
            acc[nt] = __builtin_amdgcn_wmma_f32_16x16x32_f16(
                false, a.h, false, bf[nt].h, (short)0, acc[nt], false, false);
        }
    }

    // epilogue: bias + relu -> Xh1 (this wave's own 16 rows)
    #pragma unroll
    for (int nt = 0; nt < 8; ++nt) {
        const int col = nt * 16 + m;
        const float bias = b1s[col];
        #pragma unroll
        for (int i = 0; i < 8; ++i) {
            float v = acc[nt][i] + bias;
            v = v > 0.f ? v : 0.f;
            const int row = 16 * w + hi * 8 + i;   // C layout: VGPR i -> M=i(+8)
            Xh1[row * XH_STRIDE + col] = (_Float16)v;
        }
    }
    __syncthreads();

    // ---- GEMM2: h2[64x128] = relu(h1 @ W2 + b2), K = 128 = 4 tiles ----
    v8f acc2[8];
    #pragma unroll
    for (int nt = 0; nt < 8; ++nt) acc2[nt] = vzero;

    #pragma unroll
    for (int kt = 0; kt < 4; ++kt) {
        Frag a;
        const unsigned* ap = XhU + rowA * (XH_STRIDE / 2) + kt * 16 + hi * 4;
        a.q[0] = *(const u32x4*)(ap);
        a.q[1] = *(const u32x4*)(ap + 8);
        Frag bf[8];
        #pragma unroll
        for (int nt = 0; nt < 8; ++nt) {
            const unsigned* bp = w2pt + (size_t)(nt * 16 + m) * KP2 + kt * 16 + hi * 4;
            bf[nt].q[0] = *(const u32x4*)(bp);
            bf[nt].q[1] = *(const u32x4*)(bp + 8);
        }
        __builtin_amdgcn_sched_barrier(0);
        #pragma unroll
        for (int nt = 0; nt < 8; ++nt) {
            acc2[nt] = __builtin_amdgcn_wmma_f32_16x16x32_f16(
                false, a.h, false, bf[nt].h, (short)0, acc2[nt], false, false);
        }
    }

    // ---- GEMM3 (N=1) in registers: logit = relu(h2) @ W3 + b3 ----
    float part[8];
    #pragma unroll
    for (int i = 0; i < 8; ++i) part[i] = 0.f;
    #pragma unroll
    for (int nt = 0; nt < 8; ++nt) {
        const int col = nt * 16 + m;
        const float bc = b2s[col];
        const float wc = W3s[col];
        #pragma unroll
        for (int i = 0; i < 8; ++i) {
            float v = acc2[nt][i] + bc;
            v = v > 0.f ? v : 0.f;
            part[i] += v * wc;
        }
    }
    const float b3v = b3[0];
    #pragma unroll
    for (int i = 0; i < 8; ++i) {
        float s = part[i];
        s += __shfl_xor(s, 1, 32);      // reduce across the 16-lane N group
        s += __shfl_xor(s, 2, 32);
        s += __shfl_xor(s, 4, 32);
        s += __shfl_xor(s, 8, 32);
        if (m == 0) {
            const int e = e0 + 16 * w + hi * 8 + i;
            if (e < nE) out[e] = s + b3v;
        }
    }
}

extern "C" void kernel_launch(void* const* d_in, const int* in_sizes, int n_in,
                              void* d_out, int out_size, void* d_ws, size_t ws_size,
                              hipStream_t stream) {
    const float* z    = (const float*)d_in[0];
    const int*   ei   = (const int*)d_in[1];   // [2][nE]
    const float* attr = (const float*)d_in[2];
    const float* lre  = (const float*)d_in[3];
    const float* W1   = (const float*)d_in[4];
    const float* b1   = (const float*)d_in[5];
    const float* W2   = (const float*)d_in[6];
    const float* b2   = (const float*)d_in[7];
    const float* W3   = (const float*)d_in[8];
    const float* b3   = (const float*)d_in[9];
    float* out = (float*)d_out;

    const int nE = in_sizes[1] / 2;

    unsigned* w1pt = (unsigned*)d_ws;                   // 128*288*4 = 147456 B
    unsigned* w2pt = w1pt + (size_t)HIDDEN * KP1;       // 128*64*4  =  32768 B

    pack_weights<<<KP1 + KP2, 128, 0, stream>>>(W1, W2, w1pt, w2pt);

    const int nBlocks = (nE + TILE_E - 1) / TILE_E;     // 15625 for 1M edges
    edge_mlp<<<nBlocks, 128, LDS_TOTAL, stream>>>(
        z, ei, ei + nE, attr, lre, b1, b2, W3, b3, w1pt, w2pt, out, nE);
}
